// SequenceGraphAtt_83880711290998
// MI455X (gfx1250) — matrix operand order repaired and checked by reading, not compiled, not run on hardware
//
#include <hip/hip_runtime.h>
#include <hip/hip_fp16.h>
#include <math.h>

typedef __attribute__((ext_vector_type(16))) _Float16 v16h;
typedef __attribute__((ext_vector_type(8)))  _Float16 v8h;
typedef __attribute__((ext_vector_type(8)))  float    v8f;
typedef __attribute__((ext_vector_type(4)))  unsigned int u32x4;
typedef __attribute__((ext_vector_type(8)))  int          i32x8;
typedef __attribute__((ext_vector_type(4)))  int          i32x4;
typedef _Float16 f16;

#define DI __device__ __forceinline__

// Problem constants (match reference)
constexpr int Bn = 1024, Tn = 64, Dn = 256, Hn = 128;
constexpr int An = 64, Cn = 3, S1n = 10, S2n = 10;
constexpr int G3 = 3 * Hn;            // 384 gate columns
constexpr int Mseq = Tn * Bn;         // 65536 rows for input GEMMs
constexpr int Mnf  = Bn * S2n * S1n;  // 102400 gathered neighbor rows

// ---------------------------------------------------------------------------
// WMMA fragment loaders (ISA 7.12.2 layouts, wave32)
// ---------------------------------------------------------------------------
DI v16h load_a_frag_g(const f16* __restrict__ A, int lda, int m0, int k0) {
  const int lane = threadIdx.x & 31;
  const f16* p = A + (size_t)(m0 + (lane & 15)) * lda + k0 + ((lane >> 4) << 3);
  v8h lo = *(const v8h*)p;
  v8h hi = *(const v8h*)(p + 16);
  v16h r;
#pragma unroll
  for (int i = 0; i < 8; ++i) { r[i] = lo[i]; r[8 + i] = hi[i]; }
  return r;
}

DI v16h load_a_frag_lds(const f16* A, int lda, int k0) {
  const int lane = threadIdx.x & 31;
  const f16* p = A + (lane & 15) * lda + k0 + ((lane >> 4) << 3);
  v16h r;
#pragma unroll
  for (int i = 0; i < 8; ++i) { r[i] = p[i]; r[8 + i] = p[16 + i]; }
  return r;
}

DI v16h load_b_frag_wT(const f16* __restrict__ W, int ldw, int k0, int n0) {
  const int lane = threadIdx.x & 31;
  const f16* p = W + (size_t)(n0 + (lane & 15)) * ldw + k0 + ((lane >> 4) << 3);
  v8h lo = *(const v8h*)p;
  v8h hi = *(const v8h*)(p + 16);
  v16h r;
#pragma unroll
  for (int i = 0; i < 8; ++i) { r[i] = lo[i]; r[8 + i] = hi[i]; }
  return r;
}

// ---------------------------------------------------------------------------
// Tensor Data Mover: 1-D contiguous copy of `nhalf` f16 elements from global
// memory into LDS at byte offset lds_off. D# per CDNA5 ISA:
//   group0: count=1, lds_addr, global_addr[56:0], type=2
//   group1: data_size=2B, tensor_dim0=tile_dim0=nhalf, dim1=1, stride=nhalf
// amdgpu-toolchain (clang-23) 6-arg builtin form. Tracked by TENSORcnt.
// ---------------------------------------------------------------------------
DI void tdm_load_1d(const f16* gsrc, unsigned int lds_off, int nhalf) {
  unsigned long long ga = (unsigned long long)(uintptr_t)gsrc;
  u32x4 g0;
  g0[0] = 1u;                                            // count = 1 (valid)
  g0[1] = lds_off;                                       // lds_addr
  g0[2] = (unsigned int)(ga & 0xFFFFFFFFu);              // global_addr lo
  g0[3] = (unsigned int)((ga >> 32) & 0x01FFFFFFu)       // global_addr hi
          | (2u << 30);                                  // type = 2 (image)
  i32x8 g1;
  g1[0] = 0x00010000;                                    // data_size = 2 bytes
  g1[1] = (int)((unsigned)(nhalf & 0xFFFF) << 16);       // tensor_dim0 lo16
  g1[2] = (int)(((unsigned)nhalf >> 16) & 0xFFFF)        // tensor_dim0 hi16
          | (1 << 16);                                   // tensor_dim1 = 1
  g1[3] = (int)((unsigned)(nhalf & 0xFFFF) << 16);       // tile_dim0
  g1[4] = 1;                                             // tile_dim1 = 1
  g1[5] = nhalf;                                         // tensor_dim0_stride
  g1[6] = 0;
  g1[7] = 0;
  i32x4 gz4 = {0, 0, 0, 0};
  i32x8 gz8 = {0, 0, 0, 0, 0, 0, 0, 0};
  __builtin_amdgcn_tensor_load_to_lds(g0, g1, gz4, gz4, gz8, 0);
}

// ---------------------------------------------------------------------------
// Weight conversion f32 -> f16
// ---------------------------------------------------------------------------
__global__ void cvt_f32_f16_kernel(const float* __restrict__ src,
                                   f16* __restrict__ dst, int n) {
  int g = blockIdx.x * blockDim.x + threadIdx.x;
  if (g < n) dst[g] = (f16)src[g];
}

// ---------------------------------------------------------------------------
// Embedding gather: X[t*B+b, :] = (f16)emb_table[seq_input[b, t], :]
// ---------------------------------------------------------------------------
__global__ void embed_gather_kernel(const int* __restrict__ seq,
                                    const float* __restrict__ emb,
                                    f16* __restrict__ X) {
  int g = blockIdx.x * blockDim.x + threadIdx.x;
  int chunk = g & 31;            // 32 chunks of 8 elems = 256
  int rowi  = g >> 5;            // t*B + b
  if (rowi >= Mseq) return;
  int t = rowi >> 10;            // / B(=1024)
  int b = rowi & 1023;
  int idx = seq[b * Tn + t];
  const float* s = emb + (size_t)idx * Dn + chunk * 8;
  f16* d = X + (size_t)rowi * Dn + chunk * 8;
#pragma unroll
  for (int i = 0; i < 8; ++i) d[i] = (f16)s[i];
}

// Generic row gather: out[r, :] = (f16)table[idx[r], :], D=256
__global__ void row_gather_kernel(const int* __restrict__ idx,
                                  const float* __restrict__ table,
                                  f16* __restrict__ out, int rows) {
  int g = blockIdx.x * blockDim.x + threadIdx.x;
  int chunk = g & 31;
  int r = g >> 5;
  if (r >= rows) return;
  const float* s = table + (size_t)idx[r] * Dn + chunk * 8;
  f16* d = out + (size_t)r * Dn + chunk * 8;
#pragma unroll
  for (int i = 0; i < 8; ++i) d[i] = (f16)s[i];
}

// ---------------------------------------------------------------------------
// WMMA GEMM: C[M,N] = act(A[M,K=256] * W[N,K]^T + bias), f16 A/W, f32 acc.
// grid = (N/16, M/64), block = 128 (4 waves, one 16x16 C tile per wave).
// ACT: 0 = none, 1 = relu, 2 = tanh. Ch/Cf nullable.
// ---------------------------------------------------------------------------
template <int ACT>
__global__ void gemm_wT_kernel(const f16* __restrict__ A, const f16* __restrict__ W,
                               const float* __restrict__ bias,
                               f16* __restrict__ Ch, float* __restrict__ Cf,
                               int M, int N) {
  constexpr int K = 256;
  const int lane = threadIdx.x & 31;
  const int wid  = threadIdx.x >> 5;
  const int n0 = blockIdx.x * 16;
  const int m0 = (blockIdx.y * 4 + wid) * 16;
  if (m0 >= M) return;

  v8f acc = {};
#pragma unroll
  for (int k0 = 0; k0 < K; k0 += 32) {
    // speculative prefetch of the next A chunk (global_prefetch_b8 path)
    __builtin_prefetch(A + (size_t)(m0 + (lane & 15)) * K + k0 + 128, 0, 1);
    v16h a = load_a_frag_g(A, K, m0, k0);
    v16h b = load_b_frag_wT(W, K, k0, n0);
    acc = __builtin_amdgcn_wmma_f32_16x16x32_f16(false, a, false, b,
                                                 (short)0, acc, false, false);
  }

  const int col   = n0 + (lane & 15);
  const int rbase = m0 + ((lane >> 4) << 3);
  const float bv = bias ? bias[col] : 0.f;
#pragma unroll
  for (int r = 0; r < 8; ++r) {
    float v = acc[r] + bv;
    if (ACT == 1) v = v > 0.f ? v : 0.f;
    if (ACT == 2) v = tanhf(v);
    size_t o = (size_t)(rbase + r) * N + col;
    if (Ch) Ch[o] = (f16)v;
    if (Cf) Cf[o] = v;
  }
}

// ---------------------------------------------------------------------------
// GRU recurrent scan (one direction, one layer).
// Each block owns 16 batch rows for all T=64 steps.
//  - h tile in LDS (f16); Whh in per-wave register WMMA B-fragments
//    (launch_bounds(256,1) gives the VGPR budget to keep them resident).
//  - gate inputs gi[t] (contiguous 16x384 f16 = 12KB) are double-buffered in
//    LDS via the Tensor Data Mover: wave 0 prefetches step t+1 while all
//    waves compute step t; sync = s_wait_tensorcnt + workgroup barrier.
// gi: [T,B,384] f16 (includes bih). Yh nullable ([T,B,256], write at ycol).
// hT nullable ([B,256] f32 final hidden, write at ycol).
// ---------------------------------------------------------------------------
__global__ void __launch_bounds__(256, 1)
gru_scan_kernel(const f16* __restrict__ gi,
                const f16* __restrict__ Whh,   // [384,128] f16
                const float* __restrict__ bhh, // [384]
                f16* __restrict__ Yh,
                float* __restrict__ hT,
                int ycol, int reverse) {
  const int lane = threadIdx.x & 31;
  const int wid  = threadIdx.x >> 5;  // 0..7
  const int tid  = threadIdx.x;
  const int b0   = blockIdx.x * 16;

  __shared__ f16   hS[16][136];        // h tile, padded
  __shared__ float ghS[16][392];       // gh tile, padded
  __shared__ f16   giS[2][16 * G3];    // TDM double buffer, 2 x 12KB

  // Preload Whh fragments into registers: wave wid -> n-tiles wid*3 + j
  v16h bw[3][4];
#pragma unroll
  for (int j = 0; j < 3; ++j)
#pragma unroll
    for (int kk = 0; kk < 4; ++kk)
      bw[j][kk] = load_b_frag_wT(Whh, Hn, kk * 32, (wid * 3 + j) * 16);

  // h0 = 0
  for (int i = tid; i < 16 * 136; i += 256) (&hS[0][0])[i] = (f16)0.f;

  const int row   = tid >> 4;        // 0..15 (batch row within tile)
  const int cbase = (tid & 15) * 8;  // 8 hidden columns per thread

  float bh_r[8], bh_z[8], bh_n[8];
#pragma unroll
  for (int e = 0; e < 8; ++e) {
    bh_r[e] = bhh[cbase + e];
    bh_z[e] = bhh[Hn + cbase + e];
    bh_n[e] = bhh[2 * Hn + cbase + e];
  }

  // Prefetch gi for the first step into buffer 0 (wave 0 only: TENSORcnt is
  // per-wave; cross-wave visibility is handled by wait + barrier below).
  const unsigned int giS_base = (unsigned int)(uintptr_t)(void*)&giS[0][0];
  if (wid == 0) {
    const int tt0 = reverse ? (Tn - 1) : 0;
    tdm_load_1d(gi + ((size_t)tt0 * Bn + b0) * G3, giS_base, 16 * G3);
  }

  for (int t = 0; t < Tn; ++t) {
    const int tt = reverse ? (Tn - 1 - t) : t;
    __syncthreads();  // prev-step gate writes to hS/giS consumers done

    // Issue TDM prefetch of step t+1 into the other buffer (clamped; the
    // redundant final issue is harmless and waited out by S_ENDPGM).
    if (wid == 0) {
      const int tn = (t + 1 < Tn) ? (t + 1) : (Tn - 1);
      const int tt2 = reverse ? (Tn - 1 - tn) : tn;
      tdm_load_1d(gi + ((size_t)tt2 * Bn + b0) * G3,
                  giS_base + (unsigned)((tn & 1) * 16 * G3 * 2), 16 * G3);
    }

    // gh = h * Whh^T  (overlaps with the in-flight TDM transfer)
#pragma unroll
    for (int j = 0; j < 3; ++j) {
      v8f acc = {};
#pragma unroll
      for (int kk = 0; kk < 4; ++kk) {
        v16h a = load_a_frag_lds(&hS[0][0], 136, kk * 32);
        acc = __builtin_amdgcn_wmma_f32_16x16x32_f16(false, a, false, bw[j][kk],
                                                     (short)0, acc, false, false);
      }
      const int nt = (wid * 3 + j) * 16;
#pragma unroll
      for (int r = 0; r < 8; ++r)
        ghS[r + ((lane >> 4) << 3)][nt + (lane & 15)] = acc[r];
    }

    // Ensure gi[t] landed (loads complete in order: <=1 outstanding leaves
    // only the t+1 prefetch in flight), then make it visible to all waves.
    if (wid == 0) __builtin_amdgcn_s_wait_tensorcnt(1);
    __syncthreads();

    // Gate math: each thread owns (row, cbase..cbase+7); gi read from LDS.
    const f16* grow = &giS[t & 1][row * G3];
    const size_t ybase = ((size_t)tt * Bn + (b0 + row)) * (size_t)(2 * Hn) + ycol;
#pragma unroll
    for (int e = 0; e < 8; ++e) {
      const int col = cbase + e;
      float i_r = (float)grow[col];
      float i_z = (float)grow[Hn + col];
      float i_n = (float)grow[2 * Hn + col];
      float h_r = ghS[row][col] + bh_r[e];
      float h_z = ghS[row][Hn + col] + bh_z[e];
      float h_n = ghS[row][2 * Hn + col] + bh_n[e];
      float rg = 1.f / (1.f + __expf(-(i_r + h_r)));
      float zg = 1.f / (1.f + __expf(-(i_z + h_z)));
      float ng = tanhf(i_n + rg * h_n);
      float hp = (float)hS[row][col];
      float hn = (1.f - zg) * ng + zg * hp;
      hS[row][col] = (f16)hn;
      if (Yh) Yh[ybase + col] = (f16)hn;
      if (hT && t == Tn - 1)
        hT[(size_t)(b0 + row) * (2 * Hn) + ycol + col] = hn;
    }
  }
}

// ---------------------------------------------------------------------------
// Attention reduce: given a = tanh(x W^T + b) in f32 [M*S, 64], compute
// s_j = a_j . v, softmax over S, out[m,:] = relu(invn * sum_j w_j x_j) (f16).
// One block (256 threads) per m; D = 256.
// ---------------------------------------------------------------------------
__global__ void att_reduce_kernel(const float* __restrict__ Aact,
                                  const float* __restrict__ v,
                                  const f16* __restrict__ X,
                                  f16* __restrict__ out,
                                  int S, float invn) {
  const int m = blockIdx.x;
  const int tid = threadIdx.x;
  __shared__ float sw[16];
  __shared__ float red[256];

  for (int j = 0; j < S; ++j) {
    float p = 0.f;
    if (tid < An) p = Aact[(size_t)(m * S + j) * An + tid] * v[tid];
    red[tid] = p;
    __syncthreads();
    for (int off = 128; off > 0; off >>= 1) {
      if (tid < off) red[tid] += red[tid + off];
      __syncthreads();
    }
    if (tid == 0) sw[j] = red[0];
    __syncthreads();
  }
  if (tid == 0) {
    float mx = sw[0];
    for (int j = 1; j < S; ++j) mx = fmaxf(mx, sw[j]);
    float sum = 0.f;
    for (int j = 0; j < S; ++j) { sw[j] = __expf(sw[j] - mx); sum += sw[j]; }
    float inv = 1.f / sum;
    for (int j = 0; j < S; ++j) sw[j] *= inv;
  }
  __syncthreads();

  float acc = 0.f;
  for (int j = 0; j < S; ++j)
    acc += sw[j] * (float)X[(size_t)(m * S + j) * Dn + tid];
  acc *= invn;
  acc = acc > 0.f ? acc : 0.f;
  out[(size_t)m * Dn + tid] = (f16)acc;
}

// ---------------------------------------------------------------------------
// Final combine: seq-attention over the 2-element {rnn, graph} pair + proj.
// One block (64 threads) per batch row.
// ---------------------------------------------------------------------------
__global__ void combine_kernel(const float* __restrict__ rnnE,
                               const float* __restrict__ graphE,
                               const float* __restrict__ W3,
                               const float* __restrict__ b3,
                               const float* __restrict__ v3,
                               const float* __restrict__ W2,
                               float* __restrict__ outp) {
  const int b = blockIdx.x;
  const int tid = threadIdx.x;  // 64
  __shared__ float red[64];
  __shared__ float sS[2];
  __shared__ float wpair[2];
  __shared__ float comb[256];

  const float* r = rnnE + (size_t)b * 256;
  const float* g = graphE + (size_t)b * 256;

  float d1 = 0.f, d2 = 0.f;
  const float* w = W3 + (size_t)tid * 256;
  for (int k = 0; k < 256; ++k) { d1 += w[k] * r[k]; d2 += w[k] * g[k]; }
  float a1 = tanhf(d1 + b3[tid]);
  float a2 = tanhf(d2 + b3[tid]);

  red[tid] = a1 * v3[tid];
  __syncthreads();
  for (int o = 32; o > 0; o >>= 1) { if (tid < o) red[tid] += red[tid + o]; __syncthreads(); }
  if (tid == 0) sS[0] = red[0];
  __syncthreads();
  red[tid] = a2 * v3[tid];
  __syncthreads();
  for (int o = 32; o > 0; o >>= 1) { if (tid < o) red[tid] += red[tid + o]; __syncthreads(); }
  if (tid == 0) sS[1] = red[0];
  __syncthreads();

  if (tid == 0) {
    float mx = fmaxf(sS[0], sS[1]);
    float e1 = __expf(sS[0] - mx), e2 = __expf(sS[1] - mx);
    float inv = 1.f / (e1 + e2);
    wpair[0] = e1 * inv; wpair[1] = e2 * inv;
  }
  __syncthreads();

#pragma unroll
  for (int i = 0; i < 4; ++i) {
    int d = tid + 64 * i;
    comb[d] = wpair[0] * r[d] + wpair[1] * g[d];
  }
  __syncthreads();
  if (tid < Cn) {
    float acc = 0.f;
    for (int d = 0; d < 256; ++d) acc += comb[d] * W2[d * Cn + tid];
    outp[(size_t)b * Cn + tid] = acc;
  }
}

// ---------------------------------------------------------------------------
// Launch
// ---------------------------------------------------------------------------
extern "C" void kernel_launch(void* const* d_in, const int* in_sizes, int n_in,
                              void* d_out, int out_size, void* d_ws, size_t ws_size,
                              hipStream_t stream) {
  (void)in_sizes; (void)n_in; (void)out_size; (void)ws_size;

  const int*   seq_input = (const int*)d_in[1];
  const int*   neigh1    = (const int*)d_in[2];
  const float* feat      = (const float*)d_in[3];
  const float* emb       = (const float*)d_in[4];
  const float* Wih       = (const float*)d_in[5];
  const float* Whh       = (const float*)d_in[6];
  const float* bih       = (const float*)d_in[7];
  const float* bhh       = (const float*)d_in[8];
  const float* att1_W    = (const float*)d_in[9];
  const float* att1_b    = (const float*)d_in[10];
  const float* att1_v    = (const float*)d_in[11];
  const float* enc1_W    = (const float*)d_in[12];
  const float* att2_W    = (const float*)d_in[13];
  const float* att2_b    = (const float*)d_in[14];
  const float* att2_v    = (const float*)d_in[15];
  const float* enc2_W    = (const float*)d_in[16];
  const float* att3_W    = (const float*)d_in[17];
  const float* att3_b    = (const float*)d_in[18];
  const float* att3_v    = (const float*)d_in[19];
  const float* weight2   = (const float*)d_in[20];
  float* outp = (float*)d_out;

  char* ws = (char*)d_ws;
  size_t off = 0;
  auto alloc = [&](size_t bytes) -> char* {
    char* p = ws + off;
    off = (off + bytes + 255) & ~(size_t)255;
    return p;
  };

  // --- reusable pool (sequence phase layout; graph phase re-carves it) ---
  const size_t szX0  = (size_t)Mseq * Dn * 2;   // 32 MB
  const size_t szGI  = (size_t)Mseq * G3 * 2;   // 48 MB
  f16* X0h = (f16*)alloc(szX0);
  f16* GIF = (f16*)alloc(szGI);
  f16* GIB = (f16*)alloc(szGI);
  f16* Y0h = (f16*)alloc(szX0);
  size_t pool_end = off;

  // --- persistent region (survives both phases) ---
  float* rnnE = (float*)alloc((size_t)Bn * 256 * 4);
  float* GEf  = (float*)alloc((size_t)Bn * 256 * 4);
  f16* wWih_h  = (f16*)alloc((size_t)2 * 2 * G3 * Dn * 2);
  f16* wWhh_h  = (f16*)alloc((size_t)2 * 2 * G3 * Hn * 2);
  f16* att1W_h = (f16*)alloc((size_t)An * Dn * 2);
  f16* enc1W_h = (f16*)alloc((size_t)256 * 256 * 2);
  f16* att2W_h = (f16*)alloc((size_t)An * 256 * 2);
  f16* enc2W_h = (f16*)alloc((size_t)256 * 256 * 2);

  // --- graph-phase aliases over the pool ---
  off = 0;
  f16*   NFh = (f16*)alloc((size_t)Mnf * Dn * 2);          // 50 MB
  float* A1  = (float*)alloc((size_t)Mnf * An * 4);        // 25 MB
  f16*   H1h = (f16*)alloc((size_t)Bn * S2n * 256 * 2);    // 5 MB
  f16*   E1h = (f16*)alloc((size_t)Bn * S2n * 256 * 2);    // 5 MB
  float* A2  = (float*)alloc((size_t)Bn * S2n * An * 4);   // 2.5 MB
  f16*   H2h = (f16*)alloc((size_t)Bn * 256 * 2);          // 0.5 MB
  (void)pool_end;

  // ---- 0. convert weights to f16 ----
  auto cvt = [&](const float* s, f16* d, int n) {
    cvt_f32_f16_kernel<<<(n + 255) / 256, 256, 0, stream>>>(s, d, n);
  };
  cvt(Wih,    wWih_h,  2 * 2 * G3 * Dn);
  cvt(Whh,    wWhh_h,  2 * 2 * G3 * Hn);
  cvt(att1_W, att1W_h, An * Dn);
  cvt(enc1_W, enc1W_h, 256 * 256);
  cvt(att2_W, att2W_h, An * 256);
  cvt(enc2_W, enc2W_h, 256 * 256);

  // ---- 1. sequence branch ----
  embed_gather_kernel<<<(Mseq * 32) / 256, 256, 0, stream>>>(seq_input, emb, X0h);

  auto wih = [&](int l, int d) { return wWih_h + (size_t)(l * 2 + d) * G3 * Dn; };
  auto whh = [&](int l, int d) { return wWhh_h + (size_t)(l * 2 + d) * G3 * Hn; };
  auto bihp = [&](int l, int d) { return bih + (size_t)(l * 2 + d) * G3; };
  auto bhhp = [&](int l, int d) { return bhh + (size_t)(l * 2 + d) * G3; };

  dim3 gGI(G3 / 16, Mseq / 64);
  // layer 0
  gemm_wT_kernel<0><<<gGI, 128, 0, stream>>>(X0h, wih(0, 0), bihp(0, 0), GIF, nullptr, Mseq, G3);
  gemm_wT_kernel<0><<<gGI, 128, 0, stream>>>(X0h, wih(0, 1), bihp(0, 1), GIB, nullptr, Mseq, G3);
  gru_scan_kernel<<<Bn / 16, 256, 0, stream>>>(GIF, whh(0, 0), bhhp(0, 0), Y0h, nullptr, 0,  0);
  gru_scan_kernel<<<Bn / 16, 256, 0, stream>>>(GIB, whh(0, 1), bhhp(0, 1), Y0h, nullptr, Hn, 1);
  // layer 1 (input = Y0h, only final hidden states needed -> rnnE)
  gemm_wT_kernel<0><<<gGI, 128, 0, stream>>>(Y0h, wih(1, 0), bihp(1, 0), GIF, nullptr, Mseq, G3);
  gemm_wT_kernel<0><<<gGI, 128, 0, stream>>>(Y0h, wih(1, 1), bihp(1, 1), GIB, nullptr, Mseq, G3);
  gru_scan_kernel<<<Bn / 16, 256, 0, stream>>>(GIF, whh(1, 0), bhhp(1, 0), nullptr, rnnE, 0,  0);
  gru_scan_kernel<<<Bn / 16, 256, 0, stream>>>(GIB, whh(1, 1), bhhp(1, 1), nullptr, rnnE, Hn, 1);

  // ---- 2. graph branch (pool is free now; aliases above) ----
  row_gather_kernel<<<(Mnf * 32) / 256, 256, 0, stream>>>(neigh1, feat, NFh, Mnf);
  gemm_wT_kernel<2><<<dim3(An / 16, Mnf / 64), 128, 0, stream>>>(
      NFh, att1W_h, att1_b, nullptr, A1, Mnf, An);
  att_reduce_kernel<<<Bn * S2n, 256, 0, stream>>>(A1, att1_v, NFh, H1h, S1n, 1.f / S1n);
  gemm_wT_kernel<1><<<dim3(256 / 16, (Bn * S2n) / 64), 128, 0, stream>>>(
      H1h, enc1W_h, nullptr, E1h, nullptr, Bn * S2n, 256);
  gemm_wT_kernel<2><<<dim3(An / 16, (Bn * S2n) / 64), 128, 0, stream>>>(
      E1h, att2W_h, att2_b, nullptr, A2, Bn * S2n, An);
  att_reduce_kernel<<<Bn, 256, 0, stream>>>(A2, att2_v, E1h, H2h, S2n, 1.f / S2n);
  gemm_wT_kernel<1><<<dim3(256 / 16, Bn / 64), 128, 0, stream>>>(
      H2h, enc2W_h, nullptr, nullptr, GEf, Bn, 256);

  // ---- 3. combine + project ----
  combine_kernel<<<Bn, 64, 0, stream>>>(rnnE, GEf, att3_W, att3_b, att3_v, weight2, outp);
}
